// InecptionGCNBlock_14594298872385
// MI455X (gfx1250) — matrix operand order repaired and checked by reading, not compiled, not run on hardware
//
#include <hip/hip_runtime.h>
#include <hip/hip_bf16.h>
#include <stdint.h>

typedef __bf16 bf16_t;
typedef bf16_t v16bf __attribute__((ext_vector_type(16)));
typedef float  v8f   __attribute__((ext_vector_type(8)));

#define HID       256
#define IN_F      500
#define KPAD_IN   512
#define NCLASS    40
#define NCLS_PAD  48
#define KPAD_CAT  1280
#define BN_EPS    1e-5f

union Frag16 { v16bf v; int4 q[2]; };
union Pack8  { int4 q; bf16_t b[8]; };

static __device__ __forceinline__ bf16_t f2bf(float f) {
  uint32_t u = __builtin_bit_cast(uint32_t, f);
  u += 0x7FFFu + ((u >> 16) & 1u);          // round-to-nearest-even
  uint16_t h = (uint16_t)(u >> 16);
  return __builtin_bit_cast(bf16_t, h);
}

static __device__ __forceinline__ void store_out(float* p, float v)  { *p = v; }
static __device__ __forceinline__ void store_out(bf16_t* p, float v) { *p = f2bf(v); }

// ---------------- prep kernels ----------------

// x [M][500] f32 -> xb [M][512] bf16 (zero-padded K)
__global__ void __launch_bounds__(256) conv_x_kernel(
    const float* __restrict__ x, bf16_t* __restrict__ xb, int M) {
  int idx = blockIdx.x * 256 + threadIdx.x;
  if (idx >= M * KPAD_IN) return;
  int r = idx >> 9;            // /512
  int c = idx & (KPAD_IN - 1);
  float v = (c < IN_F) ? x[(size_t)r * IN_F + c] : 0.0f;
  xb[idx] = f2bf(v);
}

// W [K][N] row-major f32 -> Wt [N][Kpad] bf16 (n-major, zero-padded K)
__global__ void __launch_bounds__(256) conv_w_kernel(
    const float* __restrict__ W, bf16_t* __restrict__ Wt, int K, int N, int Kpad) {
  int idx = blockIdx.x * 256 + threadIdx.x;
  if (idx >= N * Kpad) return;
  int n = idx / Kpad;
  int k = idx - n * Kpad;
  float v = (k < K) ? W[(size_t)k * N + n] : 0.0f;
  Wt[idx] = f2bf(v);
}

// Wc [1268][40] -> Wct [48][1280] bf16 with K remapped to padded concat layout:
//   k 0..499    <- rows 0..499     (x, padded to 512)
//   k 512..767  <- rows 500..755   (feat 1)
//   k 768..1023 <- rows 756..1011  (feat 2)
//   k 1024..1279<- rows 1012..1267 (feat 3)
__global__ void __launch_bounds__(256) conv_wc_kernel(
    const float* __restrict__ Wc, bf16_t* __restrict__ Wct) {
  int idx = blockIdx.x * 256 + threadIdx.x;
  if (idx >= NCLS_PAD * KPAD_CAT) return;
  int n = idx / KPAD_CAT;
  int k = idx - n * KPAD_CAT;
  int ks = -1;
  if (k < IN_F)                       ks = k;
  else if (k >= 512  && k < 768)      ks = 500  + (k - 512);
  else if (k >= 768  && k < 1024)     ks = 756  + (k - 768);
  else if (k >= 1024 && k < 1280)     ks = 1012 + (k - 1024);
  float v = (n < NCLASS && ks >= 0) ? Wc[(size_t)ks * NCLASS + n] : 0.0f;
  Wct[idx] = f2bf(v);
}

// ---------------- WMMA GEMM: C[M][256] = A[M][Kpad](bf16) * Wt^T (+bias) ----------------
// 8 waves/block; wave tile 16M x 128N (8 accumulators); block tile 64M x 256N.
template <typename OutT>
__global__ void __launch_bounds__(256) gemm_bf16_n256(
    const bf16_t* __restrict__ A, int Kpad,
    const bf16_t* __restrict__ Bt,     // [256][Kpad] n-major
    const float* __restrict__ bias,    // [256] or nullptr
    OutT* __restrict__ C, int M) {
  const int wave = threadIdx.x >> 5;
  const int lane = threadIdx.x & 31;
  const int lm   = lane & 15;
  const bool hi  = lane >= 16;
  const int mbase = blockIdx.x * 64 + (wave >> 1) * 16;
  const int nbase = (wave & 1) * 128;

  v8f acc[8];
#pragma unroll
  for (int t = 0; t < 8; ++t) {
    float bv = bias ? bias[nbase + t * 16 + lm] : 0.0f;
#pragma unroll
    for (int i = 0; i < 8; ++i) acc[t][i] = bv;
  }

  int ar = mbase + lm; if (ar >= M) ar = M - 1;
  // 16-bit A layout: lanes 0-15 hold K {0..7,16..23}; lanes 16-31 hold K {8..15,24..31}
  const bf16_t* arow = A + (size_t)ar * Kpad + (hi ? 8 : 0);
  // 16-bit B layout: lane n=lm; lanes 0-15 hold K {0..7,16..23}; lanes 16-31 hold K {8..15,24..31}
  const bf16_t* brow[8];
#pragma unroll
  for (int t = 0; t < 8; ++t)
    brow[t] = Bt + (size_t)(nbase + t * 16 + lm) * Kpad + (hi ? 8 : 0);

  for (int kk = 0; kk < Kpad; kk += 32) {
    Frag16 af;
    af.q[0] = *(const int4*)(arow + kk);
    af.q[1] = *(const int4*)(arow + kk + 16);
#pragma unroll
    for (int t = 0; t < 8; ++t) {
      Frag16 bf;
      bf.q[0] = *(const int4*)(brow[t] + kk);
      bf.q[1] = *(const int4*)(brow[t] + kk + 16);
      acc[t] = __builtin_amdgcn_wmma_f32_16x16x32_bf16(
          false, af.v, false, bf.v, (short)0, acc[t], false, false);
    }
  }

  // C layout: VGPR vg, lanes 0-15 -> (m=vg, n=lm); lanes 16-31 -> (m=vg+8, n=lm)
  const int mrow = mbase + (hi ? 8 : 0);
#pragma unroll
  for (int t = 0; t < 8; ++t) {
    const int n = nbase + t * 16 + lm;
#pragma unroll
    for (int vg = 0; vg < 8; ++vg) {
      int m = mrow + vg;
      if (m < M) store_out(C + (size_t)m * HID + n, acc[t][vg]);
    }
  }
}

// ---------------- SpMM scatter: out[dst] += w * support[src] ----------------
// one wave per edge; support rows are bf16 (halves HBM gather traffic), accumulate f32
__global__ void __launch_bounds__(256) spmm_scatter(
    const int* __restrict__ src, const int* __restrict__ dst,
    const float* __restrict__ ew,
    const bf16_t* __restrict__ support, float* __restrict__ out, int E) {
  int gid  = blockIdx.x * 256 + threadIdx.x;
  int e    = gid >> 5;
  int lane = gid & 31;
  if (e >= E) return;
  int   s = src[e], d = dst[e];
  float w = ew[e];
  Pack8 u;
  u.q = ((const int4*)(support + (size_t)s * HID))[lane];   // 8 bf16, coalesced b128
  float* ob = out + (size_t)d * HID + lane * 8;
#pragma unroll
  for (int i = 0; i < 8; ++i)
    atomicAdd(ob + i, w * (float)u.b[i]);
}

// ---------------- BatchNorm ----------------
__global__ void zero_stats(float* stats) { stats[threadIdx.x] = 0.0f; }  // <<<1,512>>>

__global__ void __launch_bounds__(256) bn_stats(
    const float* __restrict__ h, float* __restrict__ stats, int M) {
  int c = threadIdx.x;                     // channel
  float s = 0.0f, s2 = 0.0f;
  for (int r = blockIdx.x; r < M; r += gridDim.x) {
    float v = h[(size_t)r * HID + c];
    s += v; s2 += v * v;
  }
  atomicAdd(&stats[c], s);
  atomicAdd(&stats[HID + c], s2);
}

__global__ void __launch_bounds__(256) bn_apply_relu(
    const float* __restrict__ h, const float* __restrict__ stats,
    const float* __restrict__ gam, const float* __restrict__ bet,
    bf16_t* __restrict__ dest, int M) {
  int idx = blockIdx.x * 256 + threadIdx.x;
  if (idx >= M * HID) return;
  int c = idx & (HID - 1);
  float inv_n  = 1.0f / (float)M;
  float mean   = stats[c] * inv_n;
  float var    = stats[HID + c] * inv_n - mean * mean;
  float scale  = gam[c] * rsqrtf(var + BN_EPS);
  float shift  = bet[c] - mean * scale;
  float v = fmaxf(h[idx] * scale + shift, 0.0f);
  dest[idx] = f2bf(v);
}

// ---------------- classifier: out[M][40] = concat(x,f1,f2,f3) @ Wc + bc ----------------
// 8 waves/block, each wave 16 rows, 3 accumulators over 48 padded classes
__global__ void __launch_bounds__(256) classifier_bf16(
    const bf16_t* __restrict__ xb, const bf16_t* __restrict__ f0,
    const bf16_t* __restrict__ f1, const bf16_t* __restrict__ f2,
    const bf16_t* __restrict__ Wct,   // [48][1280]
    const float* __restrict__ bc, float* __restrict__ out, int M) {
  const int wave = threadIdx.x >> 5;
  const int lane = threadIdx.x & 31;
  const int lm   = lane & 15;
  const bool hi  = lane >= 16;
  const int rowbase = blockIdx.x * 128 + wave * 16;

  v8f acc[3];
#pragma unroll
  for (int t = 0; t < 3; ++t) {
    int n = t * 16 + lm;
    float bv = (n < NCLASS) ? bc[n] : 0.0f;
#pragma unroll
    for (int i = 0; i < 8; ++i) acc[t][i] = bv;
  }

  int ar = rowbase + lm; if (ar >= M) ar = M - 1;
  const bf16_t* segA[4]  = { xb, f0, f1, f2 };
  const int segKp[4]     = { KPAD_IN, HID, HID, HID };
  const int segBase[4]   = { 0, 512, 768, 1024 };

#pragma unroll
  for (int seg = 0; seg < 4; ++seg) {
    const bf16_t* arow = segA[seg] + (size_t)ar * segKp[seg] + (hi ? 8 : 0);
    for (int kk = 0; kk < segKp[seg]; kk += 32) {
      Frag16 af;
      af.q[0] = *(const int4*)(arow + kk);
      af.q[1] = *(const int4*)(arow + kk + 16);
      int kg = segBase[seg] + kk;
#pragma unroll
      for (int t = 0; t < 3; ++t) {
        const bf16_t* brow = Wct + (size_t)(t * 16 + lm) * KPAD_CAT + (hi ? 8 : 0) + kg;
        Frag16 bf;
        bf.q[0] = *(const int4*)(brow);
        bf.q[1] = *(const int4*)(brow + 16);
        acc[t] = __builtin_amdgcn_wmma_f32_16x16x32_bf16(
            false, af.v, false, bf.v, (short)0, acc[t], false, false);
      }
    }
  }

  const int mrow = rowbase + (hi ? 8 : 0);
#pragma unroll
  for (int t = 0; t < 3; ++t) {
    const int n = t * 16 + lm;
    if (n < NCLASS) {
#pragma unroll
      for (int vg = 0; vg < 8; ++vg) {
        int m = mrow + vg;
        if (m < M) out[(size_t)m * NCLASS + n] = acc[t][vg];
      }
    }
  }
}

// ---------------- host orchestration ----------------
extern "C" void kernel_launch(void* const* d_in, const int* in_sizes, int n_in,
                              void* d_out, int out_size, void* d_ws, size_t ws_size,
                              hipStream_t stream) {
  (void)n_in; (void)out_size; (void)ws_size;
  const float* x       = (const float*)d_in[0];
  const int*   esrc    = (const int*)  d_in[1];
  const int*   edst    = (const int*)  d_in[2];
  const float* ew      = (const float*)d_in[3];
  const float* W_in    = (const float*)d_in[4];
  const float* SW_in   = (const float*)d_in[5];
  const float* b_in    = (const float*)d_in[6];
  const float* g_in    = (const float*)d_in[7];
  const float* beta_in = (const float*)d_in[8];
  const float* W_h     = (const float*)d_in[9];
  const float* SW_h    = (const float*)d_in[10];
  const float* b_h     = (const float*)d_in[11];
  const float* g_h     = (const float*)d_in[12];
  const float* beta_h  = (const float*)d_in[13];
  const float* Wc      = (const float*)d_in[14];
  const float* bc      = (const float*)d_in[15];
  float* out = (float*)d_out;

  const int M = in_sizes[0] / IN_F;     // 100000
  const int E = in_sizes[1];            // 3200000

  // workspace carving (256B-aligned)
  char* p = (char*)d_ws;
  auto carve = [&](size_t bytes) -> char* {
    char* r = p; p += (bytes + 255) & ~(size_t)255; return r;
  };
  bf16_t* xb      = (bf16_t*)carve((size_t)M * KPAD_IN * 2);
  bf16_t* fb0     = (bf16_t*)carve((size_t)M * HID * 2);
  bf16_t* fb1     = (bf16_t*)carve((size_t)M * HID * 2);
  bf16_t* fb2     = (bf16_t*)carve((size_t)M * HID * 2);
  bf16_t* htmp    = (bf16_t*)carve((size_t)M * HID * 2);
  bf16_t* support = (bf16_t*)carve((size_t)M * HID * 2);
  float*  outb    = (float*) carve((size_t)M * HID * 4);
  float*  stats   = (float*) carve(512 * 4);
  bf16_t* WtIn    = (bf16_t*)carve((size_t)3 * HID * KPAD_IN * 2);
  bf16_t* SWtIn   = (bf16_t*)carve((size_t)3 * HID * KPAD_IN * 2);
  bf16_t* WtH     = (bf16_t*)carve((size_t)3 * HID * HID * 2);
  bf16_t* SWtH    = (bf16_t*)carve((size_t)3 * HID * HID * 2);
  bf16_t* Wct     = (bf16_t*)carve((size_t)NCLS_PAD * KPAD_CAT * 2);

  // ---- prep ----
  conv_x_kernel<<<(M * KPAD_IN + 255) / 256, 256, 0, stream>>>(x, xb, M);
  for (int j = 0; j < 3; ++j) {
    conv_w_kernel<<<(HID * KPAD_IN + 255) / 256, 256, 0, stream>>>(
        W_in  + (size_t)j * IN_F * HID, WtIn  + (size_t)j * HID * KPAD_IN, IN_F, HID, KPAD_IN);
    conv_w_kernel<<<(HID * KPAD_IN + 255) / 256, 256, 0, stream>>>(
        SW_in + (size_t)j * IN_F * HID, SWtIn + (size_t)j * HID * KPAD_IN, IN_F, HID, KPAD_IN);
    conv_w_kernel<<<(HID * HID + 255) / 256, 256, 0, stream>>>(
        W_h   + (size_t)j * HID * HID, WtH   + (size_t)j * HID * HID, HID, HID, HID);
    conv_w_kernel<<<(HID * HID + 255) / 256, 256, 0, stream>>>(
        SW_h  + (size_t)j * HID * HID, SWtH  + (size_t)j * HID * HID, HID, HID, HID);
  }
  conv_wc_kernel<<<(NCLS_PAD * KPAD_CAT + 255) / 256, 256, 0, stream>>>(Wc, Wct);

  // ---- one graph-conv layer ----
  auto gc = [&](const bf16_t* h, int Kpad, const bf16_t* Wt, const bf16_t* SWt,
                const float* bias, const float* gam, const float* bet, bf16_t* dest) {
    int gblk = (M + 63) / 64;
    gemm_bf16_n256<bf16_t><<<gblk, 256, 0, stream>>>(h, Kpad, Wt, nullptr, support, M); // support = h@W (bf16)
    gemm_bf16_n256<float ><<<gblk, 256, 0, stream>>>(h, Kpad, SWt, bias, outb, M);      // outb = h@SW + b
    spmm_scatter<<<(E * 32 + 255) / 256, 256, 0, stream>>>(esrc, edst, ew, support, outb, E);
    zero_stats<<<1, 512, 0, stream>>>(stats);
    bn_stats<<<512, 256, 0, stream>>>(outb, stats, M);
    bn_apply_relu<<<(M * HID + 255) / 256, 256, 0, stream>>>(outb, stats, gam, bet, dest, M);
  };

  const size_t WIN = (size_t)HID * KPAD_IN;   // per-layer in-weight (transposed) stride
  const size_t WHH = (size_t)HID * HID;

  // j = 0
  gc(xb, KPAD_IN, WtIn, SWtIn, b_in, g_in, beta_in, fb0);
  // j = 1 : gc(x) then hidden gc #0
  gc(xb, KPAD_IN, WtIn + WIN, SWtIn + WIN, b_in + HID, g_in + HID, beta_in + HID, htmp);
  gc(htmp, HID, WtH, SWtH, b_h, g_h, beta_h, fb1);
  // j = 2 : gc(x) then hidden gc #1, #2
  gc(xb, KPAD_IN, WtIn + 2 * WIN, SWtIn + 2 * WIN, b_in + 2 * HID, g_in + 2 * HID, beta_in + 2 * HID, htmp);
  gc(htmp, HID, WtH + WHH,     SWtH + WHH,     b_h + HID,     g_h + HID,     beta_h + HID,     htmp);
  gc(htmp, HID, WtH + 2 * WHH, SWtH + 2 * WHH, b_h + 2 * HID, g_h + 2 * HID, beta_h + 2 * HID, fb2);

  // ---- classifier ----
  classifier_bf16<<<(M + 127) / 128, 256, 0, stream>>>(xb, fb0, fb1, fb2, Wct, bc, out, M);
}
